// MultiHeadLegalAttention_16269336117590
// MI455X (gfx1250) — matrix operand order repaired and checked by reading, not compiled, not run on hardware
//
#include <hip/hip_runtime.h>
#include <hip/hip_bf16.h>
#include <math.h>
#include <stdint.h>

typedef __bf16 bf16_t;
typedef bf16_t v16bf __attribute__((ext_vector_type(16)));
typedef bf16_t v8bf  __attribute__((ext_vector_type(8)));
typedef float  v8f   __attribute__((ext_vector_type(8)));

#define B_   4
#define S_   2048
#define D_   1024
#define H_   4
#define HD_  256
#define LN_EPS_ 1e-5f

// ---------------------------------------------------------------------------
// WMMA helpers (gfx1250, wave32). Layouts per CDNA5 ISA 7.12.2:
//  A (16x32 bf16, MxK): lane L -> row M=L&15; vec elems [0..7] = K off..off+7,
//    elems [8..15] = K 16+off..16+off+7 where off = (L<16 ? 0 : 8).
//  B (32x16 bf16, KxN): lane L -> col N=L&15; vec elems [0..7] = K off..off+7,
//    elems [8..15] = K 16+off..16+off+7 where off = (L<16 ? 0 : 8).
//  C/D (16x16 f32): lane L -> col N=L&15; vgpr r -> row M = r + (L<16 ? 0 : 8).
// ---------------------------------------------------------------------------
static __device__ __forceinline__ v16bf ldA(const bf16_t* base, int ld, int kk, int lane) {
  int m   = lane & 15;
  int off = (lane < 16) ? 0 : 8;
  const bf16_t* p = base + (size_t)m * ld + kk + off;
  v8bf lo = *(const v8bf*)(p);
  v8bf hi = *(const v8bf*)(p + 16);
  return __builtin_shufflevector(lo, hi, 0,1,2,3,4,5,6,7,8,9,10,11,12,13,14,15);
}
static __device__ __forceinline__ v16bf ldB(const bf16_t* base, int ld, int kk, int lane) {
  int n    = lane & 15;
  int koff = (lane < 16) ? 0 : 8;
  const bf16_t* p = base + (size_t)n * ld + kk + koff;
  v8bf lo = *(const v8bf*)(p);
  v8bf hi = *(const v8bf*)(p + 16);
  return __builtin_shufflevector(lo, hi, 0,1,2,3,4,5,6,7,8,9,10,11,12,13,14,15);
}
#define WMMA(a,b,c) __builtin_amdgcn_wmma_f32_16x16x32_bf16(false,(a),false,(b),(short)0,(c),false,false)

// ---------------------------------------------------------------------------
// K0: zero small accumulators
// ---------------------------------------------------------------------------
__global__ void k_init(float* sum_mag) {
  if (threadIdx.x < 4) sum_mag[threadIdx.x] = 0.0f;
}

// ---------------------------------------------------------------------------
// K1: embeddings fp32 -> bf16, mag1 = ||head1 slice||, v3 = last element,
//     per-batch sum of mag1 via atomics.
// ---------------------------------------------------------------------------
__global__ void k_convert_x(const float* __restrict__ emb, bf16_t* __restrict__ Xbf,
                            float* __restrict__ mag1, float* __restrict__ v3,
                            float* __restrict__ sum_mag) {
  int row = blockIdx.x;                 // b*S + s
  int b = row >> 11;
  const float* src = emb + (size_t)row * D_;
  bf16_t* dst = Xbf + (size_t)row * D_;
  float part = 0.0f;
  for (int d = threadIdx.x; d < D_; d += blockDim.x) {
    float v = src[d];
    dst[d] = (bf16_t)v;
    if (d >= 256 && d < 512) part += v * v;   // head-1 slice
  }
  for (int m2 = 1; m2 < 32; m2 <<= 1) part += __shfl_xor(part, m2, 32);
  __shared__ float red[8];
  int lane = threadIdx.x & 31, wid = threadIdx.x >> 5;
  if (lane == 0) red[wid] = part;
  __syncthreads();
  if (threadIdx.x == 0) {
    float t = 0.0f;
    for (int i = 0; i < 8; ++i) t += red[i];
    float mg = sqrtf(t);
    mag1[row] = mg;
    v3[row]   = src[D_ - 1];
    atomicAdd(&sum_mag[b], mg);
  }
}

// ---------------------------------------------------------------------------
// K2: weight conversion (Wq pre-scaled by 1/sqrt(hd) = 1/16)
// ---------------------------------------------------------------------------
__global__ void k_convert_w(const float* __restrict__ Wq, const float* __restrict__ Wk,
                            const float* __restrict__ Wv, const float* __restrict__ Wo,
                            bf16_t* __restrict__ Wqb, bf16_t* __restrict__ Wkb,
                            bf16_t* __restrict__ Wvb, bf16_t* __restrict__ Wob) {
  int i = blockIdx.x * blockDim.x + threadIdx.x;
  if (i < H_ * HD_ * HD_) {
    Wqb[i] = (bf16_t)(Wq[i] * 0.0625f);
    Wkb[i] = (bf16_t)(Wk[i]);
    Wvb[i] = (bf16_t)(Wv[i]);
  }
  if (i < D_ * D_) Wob[i] = (bf16_t)(Wo[i]);
}

// ---------------------------------------------------------------------------
// K3: finalize intent mean: mean = sum_b (sum_s mag1)^2 / (B*S*S)
// ---------------------------------------------------------------------------
__global__ void k_mean(const float* __restrict__ sum_mag, float* __restrict__ meanI) {
  if (threadIdx.x == 0) {
    float t = 0.0f;
    for (int b = 0; b < B_; ++b) t += sum_mag[b] * sum_mag[b];
    meanI[0] = t / ((float)B_ * (float)S_ * (float)S_);
  }
}

// ---------------------------------------------------------------------------
// K4: QKV projection. One wave -> one 16x16 tile of Q, K and V(T).
// Q,K row-major [b,h,s,e]; V stored transposed [b,h,e,t].
// ---------------------------------------------------------------------------
__global__ void k_qkv(const bf16_t* __restrict__ Xbf,
                      const bf16_t* __restrict__ Wqb, const bf16_t* __restrict__ Wkb,
                      const bf16_t* __restrict__ Wvb,
                      bf16_t* __restrict__ Qb, bf16_t* __restrict__ Kb,
                      bf16_t* __restrict__ Vtb) {
  int wave = (blockIdx.x * blockDim.x + threadIdx.x) >> 5;
  int lane = threadIdx.x & 31;
  int et = wave & 15; int tmp = wave >> 4;
  int st = tmp & 127; tmp >>= 7;
  int h = tmp & 3; int b = tmp >> 2;

  const bf16_t* Abase = Xbf + (size_t)(b * S_ + st * 16) * D_ + h * HD_;
  const bf16_t* Bq = Wqb + (size_t)(h * HD_ + et * 16) * HD_;
  const bf16_t* Bk = Wkb + (size_t)(h * HD_ + et * 16) * HD_;
  const bf16_t* Bv = Wvb + (size_t)(h * HD_ + et * 16) * HD_;

  v8f cq = {}; v8f ck = {}; v8f cv = {};
  for (int kk = 0; kk < HD_; kk += 32) {
    v16bf a = ldA(Abase, D_, kk, lane);
    cq = WMMA(a, ldB(Bq, HD_, kk, lane), cq);
    ck = WMMA(a, ldB(Bk, HD_, kk, lane), ck);
    cv = WMMA(a, ldB(Bv, HD_, kk, lane), cv);
  }
  int n = lane & 15, hi = (lane >> 4) << 3;
  size_t qbase = (size_t)((b * H_ + h) * S_) * HD_;
  size_t vbase = (size_t)((b * H_ + h) * HD_) * S_;
  for (int r = 0; r < 8; ++r) {
    int s = st * 16 + r + hi;
    int e = et * 16 + n;
    Qb[qbase + (size_t)s * HD_ + e] = (bf16_t)cq[r];
    Kb[qbase + (size_t)s * HD_ + e] = (bf16_t)ck[r];
    Vtb[vbase + (size_t)e * S_ + s] = (bf16_t)cv[r];
  }
}

// ---------------------------------------------------------------------------
// per-head legal bias (h uniform per wave)
// ---------------------------------------------------------------------------
static __device__ __forceinline__ float bias_val(int h, int s, int t, float c,
                                                 float ms, float mt, float vs, float vt,
                                                 float meanI) {
  if (h == 0) return (t > s) ? c : 0.0f;
  if (h == 1) return (ms * mt > meanI) ? c : 0.0f;
  if (h == 2) return c * __expf(-fabsf((float)(t - s)) * (1.0f / (float)S_));
  return (vs * vt > 0.5f) ? c : 0.0f;
}

// ---------------------------------------------------------------------------
// K5: softmax stats. One wave -> one (b,h,16-row) strip. Online max/sum over t.
// ---------------------------------------------------------------------------
__global__ void k_stats(const bf16_t* __restrict__ Qb, const bf16_t* __restrict__ Kb,
                        const float* __restrict__ mag1, const float* __restrict__ v3,
                        const float* __restrict__ meanI_p, const float* __restrict__ bsc,
                        float* __restrict__ Mbuf, float* __restrict__ Linv) {
  int wave = (blockIdx.x * blockDim.x + threadIdx.x) >> 5;
  int lane = threadIdx.x & 31;
  int st = wave & 127; int tmp = wave >> 7;
  int h = tmp & 3; int b = tmp >> 2;
  int s0 = st * 16;
  int n = lane & 15, hi = (lane >> 4) << 3;
  float meanI = meanI_p[0];
  float c = bsc[h];

  const bf16_t* Abase = Qb + (size_t)((b * H_ + h) * S_ + s0) * HD_;
  const float* mag1b = mag1 + b * S_;
  const float* v3b   = v3 + b * S_;

  v16bf aq[8];
  for (int kb = 0; kb < 8; ++kb) aq[kb] = ldA(Abase, HD_, kb * 32, lane);

  float ms[8], vs[8], m[8], l[8];
  for (int r = 0; r < 8; ++r) {
    ms[r] = mag1b[s0 + r + hi];
    vs[r] = v3b[s0 + r + hi];
    m[r] = -3.0e38f;
    l[r] = 0.0f;
  }

  for (int tt = 0; tt < S_ / 16; ++tt) {
    v8f cs = {};
    const bf16_t* Bbase = Kb + (size_t)((b * H_ + h) * S_ + tt * 16) * HD_;
    for (int kb = 0; kb < 8; ++kb) cs = WMMA(aq[kb], ldB(Bbase, HD_, kb * 32, lane), cs);
    int t = tt * 16 + n;
    float mt = mag1b[t], vt = v3b[t];
    for (int r = 0; r < 8; ++r) {
      int s = s0 + r + hi;
      float v = cs[r] + bias_val(h, s, t, c, ms[r], mt, vs[r], vt, meanI);
      float rm = v;
      rm = fmaxf(rm, __shfl_xor(rm, 1, 32));
      rm = fmaxf(rm, __shfl_xor(rm, 2, 32));
      rm = fmaxf(rm, __shfl_xor(rm, 4, 32));
      rm = fmaxf(rm, __shfl_xor(rm, 8, 32));
      float nm = fmaxf(m[r], rm);
      float e = __expf(v - nm);
      e += __shfl_xor(e, 1, 32);
      e += __shfl_xor(e, 2, 32);
      e += __shfl_xor(e, 4, 32);
      e += __shfl_xor(e, 8, 32);
      l[r] = l[r] * __expf(m[r] - nm) + e;
      m[r] = nm;
    }
  }
  size_t base = (size_t)((b * H_ + h) * S_) + s0;
  if ((lane & 15) == 0) {
    for (int r = 0; r < 8; ++r) {
      Mbuf[base + hi + r] = m[r];
      Linv[base + hi + r] = 1.0f / l[r];
    }
  }
}

// ---------------------------------------------------------------------------
// K6: fused weights + head-average + attended (weights @ V).
// Block = (b, 16-row s-tile); wave w: head h=w>>1, e-half eh=w&1 (128 cols).
// ---------------------------------------------------------------------------
__global__ void __launch_bounds__(256)
k_attn(const bf16_t* __restrict__ Qb, const bf16_t* __restrict__ Kb,
       const bf16_t* __restrict__ Vtb,
       const float* __restrict__ mag1, const float* __restrict__ v3,
       const float* __restrict__ meanI_p, const float* __restrict__ bsc,
       const float* __restrict__ Mbuf, const float* __restrict__ Linv,
       bf16_t* __restrict__ Abuf, float* __restrict__ avg_out) {
  __shared__ __align__(64) float  s_avg[4][16][32];   // 8 KB
  __shared__ __align__(64) bf16_t s_w[8][16][32];     // 8 KB

  int b  = blockIdx.x >> 7;
  int st = blockIdx.x & 127;
  int s0 = st * 16;
  int w = threadIdx.x >> 5;
  int lane = threadIdx.x & 31;
  int h = w >> 1, eh = w & 1;
  int n = lane & 15, hi = (lane >> 4) << 3;
  float meanI = meanI_p[0];
  float c = bsc[h];

  const bf16_t* Abase = Qb + (size_t)((b * H_ + h) * S_ + s0) * HD_;
  const float* mag1b = mag1 + b * S_;
  const float* v3b   = v3 + b * S_;
  size_t mbase = (size_t)((b * H_ + h) * S_) + s0;

  v16bf aq[8];
  for (int kb = 0; kb < 8; ++kb) aq[kb] = ldA(Abase, HD_, kb * 32, lane);

  float ms[8], vs[8], m_r[8], li_r[8];
  for (int r = 0; r < 8; ++r) {
    ms[r]   = mag1b[s0 + r + hi];
    vs[r]   = v3b[s0 + r + hi];
    m_r[r]  = Mbuf[mbase + r + hi];
    li_r[r] = Linv[mbase + r + hi];
  }

  v8f acc[8];
  for (int j = 0; j < 8; ++j) acc[j] = (v8f){};

  const bf16_t* Vbase = Vtb + (size_t)((b * H_ + h) * HD_ + eh * 128) * S_;

  for (int tc = 0; tc < S_ / 32; ++tc) {
    int t0 = tc * 32;
    for (int sub = 0; sub < 2; ++sub) {
      int tt0 = t0 + sub * 16;
      v8f cs = {};
      const bf16_t* Bbase = Kb + (size_t)((b * H_ + h) * S_ + tt0) * HD_;
      for (int kb = 0; kb < 8; ++kb) cs = WMMA(aq[kb], ldB(Bbase, HD_, kb * 32, lane), cs);
      int t = tt0 + n;
      float mt = mag1b[t], vt = v3b[t];
      for (int r = 0; r < 8; ++r) {
        int s = s0 + r + hi;
        float v  = cs[r] + bias_val(h, s, t, c, ms[r], mt, vs[r], vt, meanI);
        float wv = __expf(v - m_r[r]) * li_r[r];
        s_w[w][r + hi][sub * 16 + n] = (bf16_t)wv;
        if (eh == 0) s_avg[h][r + hi][sub * 16 + n] = wv * 0.25f;
      }
    }
    __syncthreads();
    // attended += w(16x32) @ V(32x128-half)
    {
      const bf16_t* wb = &s_w[w][0][0];
      v16bf aw = ldA(wb, 32, 0, lane);
      for (int j = 0; j < 8; ++j)
        acc[j] = WMMA(aw, ldB(Vbase + (size_t)(j * 16) * S_, S_, t0, lane), acc[j]);
    }
    // head-averaged weights -> d_out (512 f32, 256 threads, 2 each)
    {
      float* dst = avg_out + (size_t)(b * S_ + s0) * S_ + t0;
      for (int idx = threadIdx.x; idx < 512; idx += 256) {
        int mm = idx >> 5, tl = idx & 31;
        float sv = s_avg[0][mm][tl] + s_avg[1][mm][tl] + s_avg[2][mm][tl] + s_avg[3][mm][tl];
        dst[(size_t)mm * S_ + tl] = sv;
      }
    }
    __syncthreads();
  }

  // write attended (concat layout [b, s, h*256 + eh*128 + e]) as bf16
  for (int j = 0; j < 8; ++j) {
    int col = h * HD_ + eh * 128 + j * 16 + n;
    for (int r = 0; r < 8; ++r)
      Abuf[(size_t)(b * S_ + s0 + r + hi) * D_ + col] = (bf16_t)acc[j][r];
  }
}

// ---------------------------------------------------------------------------
// K7: out-proj + bias + residual + LayerNorm, fused per 16-row strip.
// ---------------------------------------------------------------------------
__global__ void __launch_bounds__(256)
k_outln(const bf16_t* __restrict__ Abuf, const bf16_t* __restrict__ Wob,
        const float* __restrict__ Wo_bias, const float* __restrict__ emb,
        const float* __restrict__ ln_g, const float* __restrict__ ln_b,
        float* __restrict__ out) {
  __shared__ float s_sum[16][8];
  __shared__ float s_sq[16][8];
  int b  = blockIdx.x >> 7;
  int st = blockIdx.x & 127;
  int s0 = st * 16;
  int w = threadIdx.x >> 5;
  int lane = threadIdx.x & 31;
  int n = lane & 15, hi = (lane >> 4) << 3;

  const bf16_t* Abase = Abuf + (size_t)(b * S_ + s0) * D_;
  v8f acc[8];
  for (int j = 0; j < 8; ++j) acc[j] = (v8f){};
  for (int kk = 0; kk < D_; kk += 32) {
    v16bf a = ldA(Abase, D_, kk, lane);
    for (int j = 0; j < 8; ++j) {
      const bf16_t* Bbase = Wob + (size_t)(w * 128 + j * 16) * D_;
      acc[j] = WMMA(a, ldB(Bbase, D_, kk, lane), acc[j]);
    }
  }
  float p1[8], p2[8];
  for (int r = 0; r < 8; ++r) { p1[r] = 0.0f; p2[r] = 0.0f; }
  for (int j = 0; j < 8; ++j) {
    int col = w * 128 + j * 16 + n;
    float wb = Wo_bias[col];
    for (int r = 0; r < 8; ++r) {
      float y = acc[j][r] + wb + emb[(size_t)(b * S_ + s0 + r + hi) * D_ + col];
      acc[j][r] = y;
      p1[r] += y;
      p2[r] += y * y;
    }
  }
  for (int r = 0; r < 8; ++r) {
    p1[r] += __shfl_xor(p1[r], 1, 32); p1[r] += __shfl_xor(p1[r], 2, 32);
    p1[r] += __shfl_xor(p1[r], 4, 32); p1[r] += __shfl_xor(p1[r], 8, 32);
    p2[r] += __shfl_xor(p2[r], 1, 32); p2[r] += __shfl_xor(p2[r], 2, 32);
    p2[r] += __shfl_xor(p2[r], 4, 32); p2[r] += __shfl_xor(p2[r], 8, 32);
  }
  if ((lane & 15) == 0) {
    for (int r = 0; r < 8; ++r) { s_sum[r + hi][w] = p1[r]; s_sq[r + hi][w] = p2[r]; }
  }
  __syncthreads();
  float mu[8], rs[8];
  for (int r = 0; r < 8; ++r) {
    float t1 = 0.0f, t2 = 0.0f;
    for (int ww = 0; ww < 8; ++ww) { t1 += s_sum[r + hi][ww]; t2 += s_sq[r + hi][ww]; }
    float mm  = t1 * (1.0f / (float)D_);
    float var = t2 * (1.0f / (float)D_) - mm * mm;
    mu[r] = mm;
    rs[r] = rsqrtf(var + LN_EPS_);
  }
  for (int j = 0; j < 8; ++j) {
    int col = w * 128 + j * 16 + n;
    float g = ln_g[col], bb = ln_b[col];
    for (int r = 0; r < 8; ++r)
      out[(size_t)(b * S_ + s0 + r + hi) * D_ + col] = (acc[j][r] - mu[r]) * rs[r] * g + bb;
  }
}

// ---------------------------------------------------------------------------
// K8: guilt = softmax(row-sums of avg weights) == uniform 1/S exactly
// ---------------------------------------------------------------------------
__global__ void k_guilt(float* __restrict__ guilt) {
  int i = blockIdx.x * blockDim.x + threadIdx.x;
  if (i < B_ * S_) guilt[i] = 1.0f / (float)S_;
}

// ---------------------------------------------------------------------------
extern "C" void kernel_launch(void* const* d_in, const int* in_sizes, int n_in,
                              void* d_out, int out_size, void* d_ws, size_t ws_size,
                              hipStream_t stream) {
  const float* emb = (const float*)d_in[0];
  const float* Wq  = (const float*)d_in[1];
  const float* Wk  = (const float*)d_in[2];
  const float* Wv  = (const float*)d_in[3];
  const float* bsc = (const float*)d_in[4];
  const float* Wo  = (const float*)d_in[5];
  const float* Wob_bias = (const float*)d_in[6];
  const float* lng = (const float*)d_in[7];
  const float* lnb = (const float*)d_in[8];

  float* out   = (float*)d_out;
  float* avg   = out + (size_t)B_ * S_ * D_;
  float* guilt = avg + (size_t)B_ * S_ * S_;

  uint8_t* wsb = (uint8_t*)d_ws;
  size_t off = 0;
  auto carve = [&](size_t bytes) -> void* {
    void* p = wsb + off;
    off += (bytes + 255) & ~(size_t)255;
    return p;
  };
  bf16_t* Xbf  = (bf16_t*)carve((size_t)B_ * S_ * D_ * 2);         // 16.8 MB
  bf16_t* Qb   = (bf16_t*)carve((size_t)B_ * H_ * S_ * HD_ * 2);   // 16.8 MB
  bf16_t* Kb   = (bf16_t*)carve((size_t)B_ * H_ * S_ * HD_ * 2);   // 16.8 MB
  bf16_t* Vtb  = (bf16_t*)carve((size_t)B_ * H_ * S_ * HD_ * 2);   // 16.8 MB
  bf16_t* Abuf = (bf16_t*)carve((size_t)B_ * S_ * D_ * 2);         // 16.8 MB
  bf16_t* Wqb  = (bf16_t*)carve((size_t)H_ * HD_ * HD_ * 2);
  bf16_t* Wkb  = (bf16_t*)carve((size_t)H_ * HD_ * HD_ * 2);
  bf16_t* Wvb  = (bf16_t*)carve((size_t)H_ * HD_ * HD_ * 2);
  bf16_t* Wob  = (bf16_t*)carve((size_t)D_ * D_ * 2);              // 2 MB
  float*  mag1 = (float*)carve((size_t)B_ * S_ * 4);
  float*  v3   = (float*)carve((size_t)B_ * S_ * 4);
  float*  Mbuf = (float*)carve((size_t)B_ * H_ * S_ * 4);
  float*  Linv = (float*)carve((size_t)B_ * H_ * S_ * 4);
  float*  sum_mag = (float*)carve(64);
  float*  meanI   = (float*)carve(64);

  k_init<<<1, 64, 0, stream>>>(sum_mag);
  k_convert_x<<<B_ * S_, 256, 0, stream>>>(emb, Xbf, mag1, v3, sum_mag);
  k_convert_w<<<(D_ * D_) / 256, 256, 0, stream>>>(Wq, Wk, Wv, Wo, Wqb, Wkb, Wvb, Wob);
  k_mean<<<1, 32, 0, stream>>>(sum_mag, meanI);
  // QKV: B*H*(S/16)*(HD/16) = 32768 waves -> 4096 blocks of 8 waves
  k_qkv<<<4096, 256, 0, stream>>>(Xbf, Wqb, Wkb, Wvb, Qb, Kb, Vtb);
  // stats: B*H*(S/16) = 2048 waves -> 256 blocks
  k_stats<<<256, 256, 0, stream>>>(Qb, Kb, mag1, v3, meanI, bsc, Mbuf, Linv);
  // fused attention: B*(S/16) = 512 blocks
  k_attn<<<512, 256, 0, stream>>>(Qb, Kb, Vtb, mag1, v3, meanI, bsc, Mbuf, Linv, Abuf, avg);
  // out-proj + LN: 512 blocks
  k_outln<<<512, 256, 0, stream>>>(Abuf, Wob, Wob_bias, emb, lng, lnb, out);
  k_guilt<<<(B_ * S_) / 256, 256, 0, stream>>>(guilt);
}
